// NeuroNN_44435731644837
// MI455X (gfx1250) — matrix operand order repaired and checked by reading, not compiled, not run on hardware
//
#include <hip/hip_runtime.h>
#include <hip/hip_bf16.h>

// ---------------- model constants ----------------
#define N_NEU   6000
#define NI_CNT  1200          // NEURON_NUM - NE = 6000 - 4800
#define NPAD    6144          // K padded to multiple of 512 (8 waves x 2 chunks x 32)
#define NROWS   12000         // [W ; W*W] stacked
#define NTILES  (NROWS / 16)  // 750 output tiles of 16 rows
#define NMAX    80
#define NAVG    30
#define PI_F    3.14159265358979323846f
#define SQRT_PI_F 1.7724538509055159f

typedef _Float16 half_t;
typedef __attribute__((ext_vector_type(16))) _Float16 v16h;
typedef __attribute__((ext_vector_type(8)))  _Float16 v8h;
typedef __attribute__((ext_vector_type(8)))  float    v8f;

// ---------------- build W (fp32 math, fp16 store) ----------------
// Whf[i][j]        = W[i,j]      (i in 0..5999)
// Whf[6000+i][j]   = W[i,j]^2
__global__ __launch_bounds__(256) void build_w_kernel(
    const float* __restrict__ hp, const float* __restrict__ randu,
    half_t* __restrict__ Whf) {
  const int j = blockIdx.x * 256 + threadIdx.x;
  const int i = blockIdx.y;
  if (j >= NPAD) return;
  const size_t oW  = (size_t)i * NPAD + j;
  const size_t oW2 = (size_t)(N_NEU + i) * NPAD + j;
  if (j >= N_NEU) {  // zero K-padding
    Whf[oW]  = (half_t)0.0f;
    Whf[oW2] = (half_t)0.0f;
    return;
  }
  const int ri   = (i < NI_CNT) ? 1 : 0;
  const int rj   = (j < NI_CNT) ? 1 : 0;
  const int conn = ri + 2 * rj;                       // block id for J, P
  int idx = j - i; if (idx < 0) idx += N_NEU;         // circulant roll index
  const int connr = ri + 2 * ((idx < NI_CNT) ? 1 : 0);
  const float theta = (float)idx * ((PI_F - 0.01f) / 5999.0f);
  const float cost  = cosf(2.0f * theta) - 1.0f;
  const float wdeg  = hp[8 + connr];                  // width row of hyperparameters
  const float wr    = (PI_F / 180.0f) * wdeg;
  const float Z     = expf(cost / (4.0f * wr * wr));
  const float arg   = 32.0f * (hp[4 + conn] * Z - randu[(size_t)i * N_NEU + j]);
  const float sg    = 1.0f / (1.0f + expf(-arg));
  const float w     = hp[conn] * sg;                  // eff row of hyperparameters
  Whf[oW]  = (half_t)w;
  Whf[oW2] = (half_t)(w * w);
}

// ---------------- init: state, FF input, Gauss-Legendre nodes ----------------
__global__ __launch_bounds__(256) void init_kernel(
    const float* __restrict__ contrast, const float* __restrict__ go,
    const float* __restrict__ pref,
    float* __restrict__ input_mean, float* __restrict__ r,
    half_t* __restrict__ rhf, float* __restrict__ maxratio,
    float* __restrict__ qn, float* __restrict__ qw) {
  const int tid = blockIdx.x * blockDim.x + threadIdx.x;
  if (tid < NPAD) rhf[tid] = (half_t)0.0f;
  if (tid < N_NEU) {
    r[tid] = 0.0f;
    const float d  = go[0] - pref[tid];
    const float wr = (PI_F / 180.0f) * 30.0f;   // W_FF
    const float cg = expf((cosf(d * (PI_F / 90.0f)) - 1.0f) / (4.0f * wr * wr));
    input_mean[tid] = contrast[0] * 20.0f * cg;
  }
  if (tid < NMAX) maxratio[tid] = 0.0f;
  if (tid < 64) {  // 64-pt Gauss-Legendre via Newton on Legendre recurrence
    const int nq = 64;
    double x  = cos(3.14159265358979323846 * ((double)tid + 0.75) / (nq + 0.5));
    double dp = 1.0;
    for (int it = 0; it < 60; ++it) {
      double p0 = 1.0, p1 = x;
      for (int jj = 2; jj <= nq; ++jj) {
        double p2 = ((2.0 * jj - 1.0) * x * p1 - (jj - 1.0) * p0) / (double)jj;
        p0 = p1; p1 = p2;
      }
      dp = (double)nq * (x * p1 - p0) / (x * x - 1.0);
      x -= p1 / dp;
    }
    qn[tid] = (float)x;
    qw[tid] = (float)(2.0 / ((1.0 - x * x) * dp * dp));
  }
}

// ---------------- WMMA mat-vec: muW[0..11999] = [W;W^2] @ r ----------------
// Roles: A = r (row M=0 only, lanes 0/16 carry it), B = 16-row W tile
// (lane l holds column N=l&15, K=(l>>4)*16+e -> one contiguous 32B run/row).
// D row M=0 = outputs: VGPR0 of lanes 0..15. 8 waves/block split K
// (2 chunks per trip, 2 accumulators), LDS-reduce.
__global__ __launch_bounds__(256) void matvec_kernel(
    const half_t* __restrict__ Whf, const half_t* __restrict__ rhf,
    float* __restrict__ muW) {
  __shared__ float red[8][16];
  const int lane = threadIdx.x & 31;
  const int wave = threadIdx.x >> 5;         // 0..7
  const int hi   = lane >> 4;                // 0/1 lane-half
  const int col  = lane & 15;                // output row within tile (= N)
  const int row  = blockIdx.x * 16 + col;
  const half_t* __restrict__ Wb = Whf + (size_t)row * NPAD + hi * 16;
  const half_t* __restrict__ rq = rhf + hi * 16;   // A (M=0) K-slots for this half
  const bool isv = (col == 0);               // lanes 0 and 16 carry r

  v8f acc0 = {}, acc1 = {};
  for (int k = wave * 64; k < NPAD; k += 512) {
    const int k1 = k + 32;
    // B: this lane's 16 contiguous K-values of its W row (2 x B128)
    v16h b0 = *(const v16h*)(Wb + k);
    v16h b1 = *(const v16h*)(Wb + k1);
    // A: r chunk in matrix row M=0 only
    v16h a0 = {}, a1 = {};
    if (isv) {
      v8h l0 = *(const v8h*)(rq + k);
      v8h h0 = *(const v8h*)(rq + k + 8);
      v8h l1 = *(const v8h*)(rq + k1);
      v8h h1 = *(const v8h*)(rq + k1 + 8);
      a0 = __builtin_shufflevector(l0, h0, 0, 1, 2, 3, 4, 5, 6, 7,
                                           8, 9, 10, 11, 12, 13, 14, 15);
      a1 = __builtin_shufflevector(l1, h1, 0, 1, 2, 3, 4, 5, 6, 7,
                                           8, 9, 10, 11, 12, 13, 14, 15);
    }
    acc0 = __builtin_amdgcn_wmma_f32_16x16x32_f16(
        false, a0, false, b0, (short)0, acc0, false, false);
    acc1 = __builtin_amdgcn_wmma_f32_16x16x32_f16(
        false, a1, false, b1, (short)0, acc1, false, false);
  }
  // D[M=0][N=col] lives in VGPR0 of lanes 0..15
  if (lane < 16) red[wave][lane] = acc0[0] + acc1[0];
  __syncthreads();
  if (threadIdx.x < 16) {
    float s = 0.0f;
    #pragma unroll
    for (int wv = 0; wv < 8; ++wv) s += red[wv][threadIdx.x];
    muW[blockIdx.x * 16 + threadIdx.x] = s;
  }
}

// ---------------- Ricciardi phi + Euler update ----------------
__device__ __forceinline__ float erfcx_dev(float y) {
  y = fminf(fmaxf(y, -9.0f), 25.0f);
  const float a   = fabsf(y);
  const float a_s = fminf(a, 5.0f);
  const float sml = expf(a_s * a_s) * erfcf(a_s);
  const float a_l = fmaxf(a, 5.0f);
  const float lrg = (1.0f / (SQRT_PI_F * a_l)) * (1.0f - 0.5f / (a_l * a_l));
  const float f   = (a <= 5.0f) ? sml : lrg;
  return (y >= 0.0f) ? f : (2.0f * expf(y * y) - f);
}

__global__ __launch_bounds__(256) void update_kernel(
    const float* __restrict__ muW, const float* __restrict__ input_mean,
    float* __restrict__ r, half_t* __restrict__ rhf,
    const float* __restrict__ qn, const float* __restrict__ qw,
    float* __restrict__ maxratio, int step) {
  const int n = blockIdx.x * blockDim.x + threadIdx.x;
  if (n >= N_NEU) return;
  const float mu  = muW[n] + input_mean[n];
  const float var = muW[N_NEU + n] + 25.0f;          // SIG_EXT^2
  const float sig = sqrtf(fmaxf(var, 1e-12f));
  const float u = (20.0f - mu) / sig;                // (VT - mu)/sig
  const float l = (10.0f - mu) / sig;                // (VR - mu)/sig
  const float c = 0.5f * (u + l);
  const float h = 0.5f * (u - l);
  float integ = 0.0f;
  #pragma unroll 4
  for (int q = 0; q < 64; ++q) {
    integ += qw[q] * erfcx_dev(-(c + h * qn[q]));
  }
  integ *= h;
  const float rate = 1.0f / (1.0f + SQRT_PI_F * fmaxf(integ, 0.0f));
  const float rv = r[n];
  const float dx = (rate - rv) * 0.1f;               // T_INV * DT
  const float r2 = rv + dx;
  const float ratio = fabsf(dx) / fmaxf(1.0f, fabsf(r2));
  atomicMax((int*)(maxratio + step), __float_as_int(ratio));  // ratio >= 0
  r[n]   = r2;
  rhf[n] = (half_t)r2;
}

// ---------------- finalize: out = [r(6000), avg_conv] ----------------
__global__ __launch_bounds__(256) void finalize_kernel(
    const float* __restrict__ r, const float* __restrict__ maxratio,
    float* __restrict__ out) {
  const int tid = blockIdx.x * blockDim.x + threadIdx.x;
  if (tid < N_NEU) out[tid] = r[tid];
  if (tid == N_NEU) {
    float s = 0.0f;
    for (int q = NMAX - NAVG; q < NMAX; ++q)
      s += (maxratio[q] < 1e-5f) ? 1.0f : 0.0f;
    out[N_NEU] = s / (float)NAVG;
  }
}

extern "C" void kernel_launch(void* const* d_in, const int* in_sizes, int n_in,
                              void* d_out, int out_size, void* d_ws, size_t ws_size,
                              hipStream_t stream) {
  const float* hp       = (const float*)d_in[0];  // [3,4] hyperparameters
  const float* contrast = (const float*)d_in[1];  // [1]
  const float* go       = (const float*)d_in[2];  // [1]
  const float* pref     = (const float*)d_in[3];  // [6000]
  const float* randu    = (const float*)d_in[4];  // [6000,6000]

  char* ws = (char*)d_ws;
  size_t off = 0;
  auto carve = [&](size_t bytes) -> void* {
    void* p = ws + off;
    off += (bytes + 255) & ~(size_t)255;
    return p;
  };
  half_t* Whf        = (half_t*)carve(sizeof(half_t) * (size_t)NROWS * NPAD); // 147.5 MB < L2
  float*  muW        = (float*)carve(sizeof(float) * NROWS);
  float*  r          = (float*)carve(sizeof(float) * N_NEU);
  half_t* rhf        = (half_t*)carve(sizeof(half_t) * NPAD);
  float*  input_mean = (float*)carve(sizeof(float) * N_NEU);
  float*  maxratio   = (float*)carve(sizeof(float) * NMAX);
  float*  qn         = (float*)carve(sizeof(float) * 64);
  float*  qw         = (float*)carve(sizeof(float) * 64);

  // 1) build fp16 [W ; W^2] (fits 192 MB L2 -> iterations are L2-resident)
  build_w_kernel<<<dim3(NPAD / 256, N_NEU), 256, 0, stream>>>(hp, randu, Whf);
  // 2) init state, feed-forward input, GL quadrature
  init_kernel<<<NPAD / 256, 256, 0, stream>>>(
      contrast, go, pref, input_mean, r, rhf, maxratio, qn, qw);
  // 3) 80 Euler fixed-point steps: WMMA mat-vec + phi update
  for (int step = 0; step < NMAX; ++step) {
    matvec_kernel<<<NTILES, 256, 0, stream>>>(Whf, rhf, muW);
    update_kernel<<<(N_NEU + 255) / 256, 256, 0, stream>>>(
        muW, input_mean, r, rhf, qn, qw, maxratio, step);
  }
  // 4) emit [r, avg convergence over last 30 steps]
  finalize_kernel<<<(N_NEU + 256) / 256, 256, 0, stream>>>(r, maxratio, (float*)d_out);
}
